// CapsuleModelNew1_13099650253332
// MI455X (gfx1250) — matrix operand order repaired and checked by reading, not compiled, not run on hardware
//
#include <hip/hip_runtime.h>

typedef __attribute__((ext_vector_type(16))) _Float16 v16h;
typedef __attribute__((ext_vector_type(8)))  _Float16 v8h;
typedef __attribute__((ext_vector_type(4)))  _Float16 v4h;
typedef __attribute__((ext_vector_type(8)))  float    v8f;

#define KDIM 1280
#define NPIX 4096
#define CCAP 256

__device__ __forceinline__ v16h mk16(v8h a, v8h b) {
  return __builtin_shufflevector(a, b, 0,1,2,3,4,5,6,7,8,9,10,11,12,13,14,15);
}
__device__ __forceinline__ v8f wmma16(v16h a, v16h b, v8f c) {
  return __builtin_amdgcn_wmma_f32_16x16x32_f16(false, a, false, b, (short)0, c, false, false);
}

// ---------------------------------------------------------------------------
// Kernel 0: prebuild f16 weight tiles in workspace.
//   wsWBT[80][288]: [Wk|Wv] transposed (N-major, k-contiguous), zero padded.
//   wsQ  [32][72] : Q padded to 32 rows / 72 cols.
// ---------------------------------------------------------------------------
__global__ __launch_bounds__(256) void prep_kernel(
    const float* __restrict__ Q, const float* __restrict__ Wk,
    const float* __restrict__ Wv, _Float16* __restrict__ wsWBT,
    _Float16* __restrict__ wsQ) {
  for (int idx = threadIdx.x; idx < 80 * 288; idx += 256) {
    int j = idx / 288, kg = idx % 288;
    float v = 0.f;
    if (kg < 258) v = (j < 64) ? Wk[kg * 64 + j] : Wv[kg * 16 + (j - 64)];
    wsWBT[idx] = (_Float16)v;
  }
  for (int idx = threadIdx.x; idx < 32 * 72; idx += 256) {
    int r = idx / 72, c = idx % 72;
    wsQ[idx] = (_Float16)((r < 19 && c < 64) ? Q[r * 64 + c] : 0.f);
  }
}

// ---------------------------------------------------------------------------
// Kernel 1: am[b][hw] = sigmoid(sum_c feat[b][c][hw] * Wa[0][c] + ba[0])
// ---------------------------------------------------------------------------
__global__ __launch_bounds__(256) void act_kernel(
    const float* __restrict__ F, const float* __restrict__ Wa,
    const float* __restrict__ ba, float* __restrict__ am) {
  __shared__ float sW[KDIM];
  const int b  = blockIdx.x >> 4;
  const int p0 = (blockIdx.x & 15) << 8;
  for (int k = threadIdx.x; k < KDIM; k += 256) sW[k] = Wa[k];  // row 0 of Wa
  __syncthreads();
  const float* Fb = F + (size_t)b * KDIM * NPIX + p0 + threadIdx.x;
  float a0 = 0.f, a1 = 0.f, a2 = 0.f, a3 = 0.f;
  for (int k = 0; k < KDIM; k += 4) {
    a0 += Fb[(size_t)(k + 0) * NPIX] * sW[k + 0];
    a1 += Fb[(size_t)(k + 1) * NPIX] * sW[k + 1];
    a2 += Fb[(size_t)(k + 2) * NPIX] * sW[k + 2];
    a3 += Fb[(size_t)(k + 3) * NPIX] * sW[k + 3];
  }
  float acc = ba[0] + (a0 + a1) + (a2 + a3);
  am[b * NPIX + p0 + threadIdx.x] = 1.f / (1.f + __expf(-acc));
}

// ---------------------------------------------------------------------------
// Kernel 2: cpT[b][hw][m] = sum_k feat[b][k][hw] * Wp[m][k] + bp[m]
// WMMA GEMM per batch (M=256, N=4096, K=1280), software-pipelined staging.
// ---------------------------------------------------------------------------
__global__ __launch_bounds__(256) void conv_kernel(
    const float* __restrict__ F, const float* __restrict__ Wp,
    const float* __restrict__ bp, float* __restrict__ cpT) {
  __shared__ __align__(16) _Float16 sA[128 * 40];  // Wp tile [m][k]
  __shared__ __align__(16) _Float16 sB[128 * 40];  // feat tile transposed [n][k]
  __shared__ float sBias[128];

  const int tid = threadIdx.x;
  const int wave = tid >> 5, lane = tid & 31;
  const int hi = lane >> 4, l15 = lane & 15;
  const int b  = blockIdx.z;
  const int M0 = blockIdx.y * 128;
  const int N0 = blockIdx.x * 128;
  const float* Fb = F + (size_t)b * KDIM * NPIX;

  if (tid < 128) sBias[tid] = bp[M0 + tid];

  const int rowA = tid >> 1, colA = (tid & 1) * 16;  // A: 16 k-contig floats/thread
  const int krB  = tid >> 3, ncB  = (tid & 7) * 16;  // B: 16 n-contig floats/thread

  float4 ra[4], rb[4];
  auto loadAB = [&](int k0) {
    const float4* pa = (const float4*)&Wp[(size_t)(M0 + rowA) * KDIM + k0 + colA];
#pragma unroll
    for (int i = 0; i < 4; ++i) ra[i] = pa[i];
    const float4* pb = (const float4*)&Fb[(size_t)(k0 + krB) * NPIX + N0 + ncB];
#pragma unroll
    for (int i = 0; i < 4; ++i) rb[i] = pb[i];
  };
  auto storeAB = [&]() {
#pragma unroll
    for (int i = 0; i < 4; ++i) {
      v4h h = {(_Float16)ra[i].x, (_Float16)ra[i].y, (_Float16)ra[i].z, (_Float16)ra[i].w};
      *(v4h*)&sA[rowA * 40 + colA + i * 4] = h;
    }
#pragma unroll
    for (int i = 0; i < 4; ++i) {           // transpose scatter: sB[n][k]
      sB[(ncB + i * 4 + 0) * 40 + krB] = (_Float16)rb[i].x;
      sB[(ncB + i * 4 + 1) * 40 + krB] = (_Float16)rb[i].y;
      sB[(ncB + i * 4 + 2) * 40 + krB] = (_Float16)rb[i].z;
      sB[(ncB + i * 4 + 3) * 40 + krB] = (_Float16)rb[i].w;
    }
  };

  v8f acc[8];
  const v8f vz = {0.f,0.f,0.f,0.f,0.f,0.f,0.f,0.f};
#pragma unroll
  for (int j = 0; j < 8; ++j) acc[j] = vz;

  loadAB(0);
  for (int k0 = 0; k0 < KDIM; k0 += 32) {
    __syncthreads();                 // previous compute done, LDS reusable
    storeAB();
    __syncthreads();
    if (k0 + 32 < KDIM) loadAB(k0 + 32);   // overlap global loads with WMMA
    // preload ALL fragments first so ds reads pipeline across the WMMA chain
    const int arow = wave * 16 + l15;
    v8h a0 = *(const v8h*)&sA[arow * 40 + hi * 8];
    v8h a1 = *(const v8h*)&sA[arow * 40 + 16 + hi * 8];
    v8h b0[8], b1[8];
#pragma unroll
    for (int j = 0; j < 8; ++j) {
      int brow = j * 16 + l15;
      b0[j] = *(const v8h*)&sB[brow * 40 + hi * 8];
      b1[j] = *(const v8h*)&sB[brow * 40 + 16 + hi * 8];
    }
    v16h af = mk16(a0, a1);
#pragma unroll
    for (int j = 0; j < 8; ++j) acc[j] = wmma16(af, mk16(b0[j], b1[j]), acc[j]);
  }
  // epilogue: C element (m = r + hi*8, n = l15) -> cpT[b][n][m], 8 m-contiguous
  const int mloc = wave * 16 + hi * 8;
#pragma unroll
  for (int j = 0; j < 8; ++j) {
    int nn = N0 + j * 16 + l15;
    float* dst = cpT + ((size_t)b * NPIX + nn) * CCAP + M0 + mloc;
#pragma unroll
    for (int r = 0; r < 8; ++r) dst[r] = acc[j][r] + sBias[mloc + r];
  }
}

// ---------------------------------------------------------------------------
// Kernel 3: routing. One block per (b, instance). 256 threads (8 wave32).
// ---------------------------------------------------------------------------
#define SM_BITMAP 0
#define SM_KEYS   512
#define SM_RELY   1536
#define SM_RELX   2560
#define SM_LOGACT 3584
#define SM_BKV    4608
#define SM_WL     4928
#define SM_MISC   4992
#define SM_Q      5008     // half [32][72]
#define SM_P      9616     // half [256][40]  poses k-chunk
#define SM_WB     30096    // half [80][40]   Wk|Wv chunk, N-major
#define SM_K      36496    // half [256][72]  kmat (B-frag friendly)
#define SM_VT     73360    // half [16][264]  vmat transposed
#define SM_S      81808    // float [32][256] scores
#define SM_ATT    114576   // half [32][264]  attn
#define SM_OP     131472   // float [32][16]  out_poses
#define SM_TOTAL  133520

__global__ __launch_bounds__(256) void route_kernel(
    const float* __restrict__ cpT, const float* __restrict__ am,
    const _Float16* __restrict__ wsWBT, const _Float16* __restrict__ wsQ,
    const float* __restrict__ bk, const float* __restrict__ bv,
    const float* __restrict__ Wl, const float* __restrict__ bl,
    const int* __restrict__ pts, float* __restrict__ out) {
  extern __shared__ char smem[];
  unsigned* bitmap = (unsigned*)(smem + SM_BITMAP);
  int*      keys   = (int*)(smem + SM_KEYS);
  float*    rely   = (float*)(smem + SM_RELY);
  float*    relx   = (float*)(smem + SM_RELX);
  float*    logact = (float*)(smem + SM_LOGACT);
  float*    bkv    = (float*)(smem + SM_BKV);
  float*    wl16   = (float*)(smem + SM_WL);
  int*      misc   = (int*)(smem + SM_MISC);
  _Float16* sQ  = (_Float16*)(smem + SM_Q);
  _Float16* sP  = (_Float16*)(smem + SM_P);
  _Float16* sWB = (_Float16*)(smem + SM_WB);
  _Float16* sK  = (_Float16*)(smem + SM_K);
  _Float16* sVT = (_Float16*)(smem + SM_VT);
  float*    sS  = (float*)(smem + SM_S);
  _Float16* sAt = (_Float16*)(smem + SM_ATT);
  float*    sOP = (float*)(smem + SM_OP);

  const int tid = threadIdx.x, wave = tid >> 5, lane = tid & 31;
  const int hi = lane >> 4, l15 = lane & 15;
  const int b = blockIdx.y, inst = blockIdx.x;

  // ---- Phase 1: dedup keys, relative positions, activations -------------
  if (tid < 128) bitmap[tid] = 0u;
  if (tid == 0) { misc[0] = 0; misc[1] = 0; misc[2] = 0; }
  if (tid < 64) bkv[tid] = bk[tid];
  else if (tid < 80) bkv[tid] = bv[tid - 64];
  if (tid < 16) wl16[tid] = Wl[tid];
  for (int idx = tid; idx < 288; idx += 256)            // Q: straight v8h copy
    *(v8h*)&sQ[idx * 8] = *(const v8h*)&wsQ[idx * 8];
  __syncthreads();
  const int* pb = pts + (size_t)(b * 32 + inst) * 2 * 256;
  {
    int py = pb[tid], px = pb[256 + tid];
    int key = ((py >> 4) << 6) | (px >> 4);
    unsigned bit = 1u << (key & 31);
    unsigned old = atomicOr(&bitmap[key >> 5], bit);
    if (!(old & bit)) { int s = atomicAdd(&misc[0], 1); keys[s] = key; }
  }
  __syncthreads();
  const int n = misc[0];
  if (tid < n) { atomicAdd(&misc[1], keys[tid] >> 6); atomicAdd(&misc[2], keys[tid] & 63); }
  __syncthreads();
  {
    float invn = 1.f / (float)(n < 1 ? 1 : n);
    if (tid < n) {
      int k = keys[tid];
      rely[tid] = ((float)(k >> 6) - (float)misc[1] * invn) * (1.f / 64.f);
      relx[tid] = ((float)(k & 63) - (float)misc[2] * invn) * (1.f / 64.f);
      logact[tid] = __logf(am[b * NPIX + k] + 1e-6f);
    } else {
      rely[tid] = 0.f; relx[tid] = 0.f; logact[tid] = -1e9f;
    }
  }

  // ---- Phase 2: [kmat|vmat] = poses[256x288] @ [Wk|Wv][288x80] ----------
  // thread == point: 8x float4 register-batched gather, pipelined with WMMA
  float4 rp[8];
  v8h rwb0 = {}, rwb1 = {};
  const int jwb = tid >> 2, cwb = (tid & 3) * 8;
  auto loadP = [&](int kc) {
    if (kc < 8) {
      if (tid < n) {
        const float4* src = (const float4*)&cpT[((size_t)b * NPIX + keys[tid]) * CCAP + kc * 32];
#pragma unroll
        for (int i = 0; i < 8; ++i) rp[i] = src[i];
      } else {
        const float4 z = make_float4(0.f, 0.f, 0.f, 0.f);
#pragma unroll
        for (int i = 0; i < 8; ++i) rp[i] = z;
      }
    } else {                                   // k rows 256..287: rel enc + zero pad
      const float4 z = make_float4(0.f, 0.f, 0.f, 0.f);
#pragma unroll
      for (int i = 0; i < 8; ++i) rp[i] = z;
      rp[0].x = rely[tid]; rp[0].y = relx[tid];
    }
  };
  auto storeP = [&]() {
#pragma unroll
    for (int i = 0; i < 8; ++i) {
      v4h h = {(_Float16)rp[i].x, (_Float16)rp[i].y, (_Float16)rp[i].z, (_Float16)rp[i].w};
      *(v4h*)&sP[tid * 40 + i * 4] = h;
    }
  };
  auto loadWB = [&](int k0) {
    rwb0 = *(const v8h*)&wsWBT[jwb * 288 + k0 + cwb];
    if (tid < 64) rwb1 = *(const v8h*)&wsWBT[(64 + jwb) * 288 + k0 + cwb];
  };
  auto storeWB = [&]() {
    *(v8h*)&sWB[jwb * 40 + cwb] = rwb0;
    if (tid < 64) *(v8h*)&sWB[(64 + jwb) * 40 + cwb] = rwb1;
  };

  const v8f vz = {0.f,0.f,0.f,0.f,0.f,0.f,0.f,0.f};
  v8f accKV[2][5];
#pragma unroll
  for (int i = 0; i < 2; ++i)
#pragma unroll
    for (int j = 0; j < 5; ++j) accKV[i][j] = vz;

  loadP(0); loadWB(0);
  for (int kc = 0; kc < 9; ++kc) {
    __syncthreads();
    storeP(); storeWB();
    __syncthreads();
    if (kc < 8) { loadP(kc + 1); loadWB((kc + 1) * 32); }  // overlap with WMMA
    // preload all fragments, then run the WMMA chain
    v8h pa0[2], pa1[2], wb0[5], wb1[5];
#pragma unroll
    for (int mtl = 0; mtl < 2; ++mtl) {
      int arow = (2 * wave + mtl) * 16 + l15;
      pa0[mtl] = *(const v8h*)&sP[arow * 40 + hi * 8];
      pa1[mtl] = *(const v8h*)&sP[arow * 40 + 16 + hi * 8];
    }
#pragma unroll
    for (int nt = 0; nt < 5; ++nt) {
      int brow = nt * 16 + l15;
      wb0[nt] = *(const v8h*)&sWB[brow * 40 + hi * 8];
      wb1[nt] = *(const v8h*)&sWB[brow * 40 + 16 + hi * 8];
    }
#pragma unroll
    for (int mtl = 0; mtl < 2; ++mtl) {
      v16h af = mk16(pa0[mtl], pa1[mtl]);
#pragma unroll
      for (int nt = 0; nt < 5; ++nt)
        accKV[mtl][nt] = wmma16(af, mk16(wb0[nt], wb1[nt]), accKV[mtl][nt]);
    }
  }
  __syncthreads();
  // epilogue: kmat -> sK[pt][dk] (half), vmat -> sVT[j][pt] (transposed)
#pragma unroll
  for (int mtl = 0; mtl < 2; ++mtl) {
    int mbase = (2 * wave + mtl) * 16 + hi * 8;
#pragma unroll
    for (int nt = 0; nt < 5; ++nt) {
      int nc = nt * 16 + l15;
#pragma unroll
      for (int r = 0; r < 8; ++r) {
        float v = accKV[mtl][nt][r] + bkv[nc];
        int m = mbase + r;
        if (nc < 64) sK[m * 72 + nc] = (_Float16)v;
        else         sVT[(nc - 64) * 264 + m] = (_Float16)v;
      }
    }
  }
  __syncthreads();

  // ---- Phase 3: scores = Qpad[32x64] @ kmat^T -> [32 caps][256 pts] -----
  v8f accS[2][2];
#pragma unroll
  for (int i = 0; i < 2; ++i) { accS[i][0] = vz; accS[i][1] = vz; }
  {
    v8h qa0[2][2], qa1[2][2], kb0[2][2], kb1[2][2];
#pragma unroll
    for (int kc = 0; kc < 2; ++kc) {
      const int k0 = kc * 32;
#pragma unroll
      for (int mt = 0; mt < 2; ++mt) {
        int arow = mt * 16 + l15;
        qa0[kc][mt] = *(const v8h*)&sQ[arow * 72 + k0 + hi * 8];
        qa1[kc][mt] = *(const v8h*)&sQ[arow * 72 + k0 + 16 + hi * 8];
      }
#pragma unroll
      for (int ntl = 0; ntl < 2; ++ntl) {
        int prow = (2 * wave + ntl) * 16 + l15;
        kb0[kc][ntl] = *(const v8h*)&sK[prow * 72 + k0 + hi * 8];
        kb1[kc][ntl] = *(const v8h*)&sK[prow * 72 + k0 + 16 + hi * 8];
      }
    }
#pragma unroll
    for (int kc = 0; kc < 2; ++kc)
#pragma unroll
      for (int mt = 0; mt < 2; ++mt) {
        v16h af = mk16(qa0[kc][mt], qa1[kc][mt]);
#pragma unroll
        for (int ntl = 0; ntl < 2; ++ntl)
          accS[mt][ntl] = wmma16(af, mk16(kb0[kc][ntl], kb1[kc][ntl]), accS[mt][ntl]);
      }
  }
#pragma unroll
  for (int mt = 0; mt < 2; ++mt) {
    int mbase = mt * 16 + hi * 8;
#pragma unroll
    for (int ntl = 0; ntl < 2; ++ntl) {
      int pt = (2 * wave + ntl) * 16 + l15;
#pragma unroll
      for (int r = 0; r < 8; ++r) sS[(mbase + r) * 256 + pt] = accS[mt][ntl][r];
    }
  }
  __syncthreads();

  // ---- Phase 4: softmax over points per capsule row ---------------------
  for (int m = wave; m < 32; m += 8) {
    float x[8], e[8], mx = -3.4e38f;
#pragma unroll
    for (int jj = 0; jj < 8; ++jj) {
      int j = lane + jj * 32;
      x[jj] = sS[m * 256 + j] * 0.125f + logact[j];   // logact=-1e9 masks invalid
      mx = fmaxf(mx, x[jj]);
    }
    for (int off = 16; off; off >>= 1) mx = fmaxf(mx, __shfl_xor(mx, off, 32));
    float sum = 0.f;
#pragma unroll
    for (int jj = 0; jj < 8; ++jj) { e[jj] = __expf(x[jj] - mx); sum += e[jj]; }
    for (int off = 16; off; off >>= 1) sum += __shfl_xor(sum, off, 32);
    float inv = 1.f / sum;
#pragma unroll
    for (int jj = 0; jj < 8; ++jj) sAt[m * 264 + lane + jj * 32] = (_Float16)(e[jj] * inv);
  }
  __syncthreads();

  // ---- Phase 5: out_poses = attn[32x256] @ vmat[256x16] -----------------
  if (wave < 2) {
    v8f accO = vz;
    const int arow = wave * 16 + l15;
#pragma unroll
    for (int kc = 0; kc < 8; kc += 2) {       // preload 2 k-steps of frags
      v8h aa0[2], aa1[2], vb0[2], vb1[2];
#pragma unroll
      for (int u = 0; u < 2; ++u) {
        const int k0 = (kc + u) * 32;
        aa0[u] = *(const v8h*)&sAt[arow * 264 + k0 + hi * 8];
        aa1[u] = *(const v8h*)&sAt[arow * 264 + k0 + 16 + hi * 8];
        vb0[u] = *(const v8h*)&sVT[l15 * 264 + k0 + hi * 8];
        vb1[u] = *(const v8h*)&sVT[l15 * 264 + k0 + 16 + hi * 8];
      }
#pragma unroll
      for (int u = 0; u < 2; ++u)
        accO = wmma16(mk16(aa0[u], aa1[u]), mk16(vb0[u], vb1[u]), accO);
    }
    int mbase = wave * 16 + hi * 8;
#pragma unroll
    for (int r = 0; r < 8; ++r) sOP[(mbase + r) * 16 + l15] = accO[r];
  }
  __syncthreads();

  // ---- Phase 6: out_acts = sigmoid(out_poses @ Wl + bl) -----------------
  if (tid < 19) {
    float acc = bl[0];
#pragma unroll
    for (int j = 0; j < 16; ++j) acc += sOP[tid * 16 + j] * wl16[j];
    out[(size_t)(b * 32 + inst) * 19 + tid] = 1.f / (1.f + __expf(-acc));
  }
}

// ---------------------------------------------------------------------------
extern "C" void kernel_launch(void* const* d_in, const int* in_sizes, int n_in,
                              void* d_out, int out_size, void* d_ws, size_t ws_size,
                              hipStream_t stream) {
  const float* feat = (const float*)d_in[0];
  const float* Wp   = (const float*)d_in[1];
  const float* bp   = (const float*)d_in[2];
  const float* Wa   = (const float*)d_in[3];
  const float* ba   = (const float*)d_in[4];
  const float* Q    = (const float*)d_in[5];
  const float* Wk   = (const float*)d_in[6];
  const float* bk   = (const float*)d_in[7];
  const float* Wv   = (const float*)d_in[8];
  const float* bv   = (const float*)d_in[9];
  const float* Wl   = (const float*)d_in[10];
  const float* bl   = (const float*)d_in[11];
  const int*   pts  = (const int*)d_in[12];
  float* out = (float*)d_out;

  char* ws = (char*)d_ws;
  float*    cpT   = (float*)ws;                                  // 32 MiB
  float*    amw   = (float*)(ws + (size_t)8 * NPIX * CCAP * 4);  // 128 KiB
  _Float16* wsWBT = (_Float16*)(ws + (size_t)8 * NPIX * CCAP * 4 + 131072);
  _Float16* wsQ   = (_Float16*)(ws + (size_t)8 * NPIX * CCAP * 4 + 131072 + 46080);

  prep_kernel<<<dim3(1), dim3(256), 0, stream>>>(Q, Wk, Wv, wsWBT, wsQ);
  act_kernel<<<dim3(128), dim3(256), 0, stream>>>(feat, Wa, ba, amw);
  conv_kernel<<<dim3(32, 2, 8), dim3(256), 0, stream>>>(feat, Wp, bp, cpT);
  route_kernel<<<dim3(32, 8), dim3(256), SM_TOTAL, stream>>>(
      cpT, amw, wsWBT, wsQ, bk, bv, Wl, bl, pts, out);
}
